// NDNRelation_83030307766794
// MI455X (gfx1250) — compile-verified, matrix operand order checked by reading, not run on hardware
//
#include <hip/hip_runtime.h>
#include <hip/hip_bf16.h>
#include <math.h>

// ---------------------------------------------------------------------------
// CDNA5 (gfx1250) scene-graph net: f16 WMMA GEMMs, f32 accumulation.
// ---------------------------------------------------------------------------
#define O_N 30000
#define T_N 100000
#define R_N 10
#define BN_EPS 0.001f
#define LEAKY_A 0.2f

typedef _Float16 half_t;
typedef __attribute__((ext_vector_type(8)))  _Float16 v8h;    // 16 bytes
typedef __attribute__((ext_vector_type(16))) _Float16 v16h;
typedef __attribute__((ext_vector_type(8)))  float    v8f;
typedef __attribute__((ext_vector_type(4)))  int      v4i;    // 16 bytes

#define GEMM_BM 128
#define GEMM_BN 128
#define GEMM_BK 32
#define LDSS    40   // LDS row stride in halves: 80B rows, 16B aligned, conflict-free

#if __has_builtin(__builtin_amdgcn_global_load_async_to_lds_b128) && \
    __has_builtin(__builtin_amdgcn_s_wait_asynccnt)
#define HAVE_ASYNC_LDS 1
typedef __attribute__((address_space(1))) v4i* as1_v4i_p;
typedef __attribute__((address_space(3))) v4i* as3_v4i_p;
#else
#define HAVE_ASYNC_LDS 0
#endif

static inline long cdiv(long a, long b) { return (a + b - 1) / b; }

__device__ inline void atomAddF(float* p, float v) { unsafeAtomicAdd(p, v); }

// ---------------------------------------------------------------------------
// TDM probe (never launched; compiled only). 6-arg clang-23 signature:
// (uint32x4 g0, int32x8 g1, int32x4 g2, int32x4 g3, int32x8 g4, i32 cpol)
// ---------------------------------------------------------------------------
#if __has_builtin(__builtin_amdgcn_tensor_load_to_lds) && \
    __has_builtin(__builtin_amdgcn_s_wait_tensorcnt)
typedef __attribute__((ext_vector_type(4))) unsigned int u32x4_t;
typedef __attribute__((ext_vector_type(8))) int          i32x8_t;
typedef __attribute__((ext_vector_type(4))) int          i32x4_t;
__global__ void k_tdm_probe(half_t* out, int flag) {
  __shared__ half_t stage[1024];
  if (flag) {
    u32x4_t g0 = {}; i32x8_t g1 = {}; i32x4_t g2 = {}; i32x4_t g3 = {}; i32x8_t g4 = {};
    g0[0] = 1u;            // D# group0: count=1
    g0[3] = (int)(2u << 30);  // D# group0: type=2 ("image")
    __builtin_amdgcn_tensor_load_to_lds(g0, g1, g2, g3, g4, 0);
    __builtin_amdgcn_s_wait_tensorcnt(0);
  }
  __syncthreads();
  out[threadIdx.x] = stage[threadIdx.x];
}
#endif

// ---------------------------------------------------------------------------
// Elementwise helpers
// ---------------------------------------------------------------------------
__global__ __launch_bounds__(256) void k_f32_to_f16(const float* __restrict__ s,
                                                    half_t* __restrict__ d, long n) {
  long stride = (long)gridDim.x * blockDim.x;
  for (long i = (long)blockIdx.x * blockDim.x + threadIdx.x; i < n; i += stride)
    d[i] = (half_t)s[i];
}

// W (K x N) f32  ->  WT (N x K) f16   (one-time per launch, coalesced reads)
__global__ __launch_bounds__(256) void k_cvt_transpose(const float* __restrict__ W,
                                                       half_t* __restrict__ WT,
                                                       int K, int N) {
  long n = (long)K * N;
  long stride = (long)gridDim.x * blockDim.x;
  for (long i = (long)blockIdx.x * blockDim.x + threadIdx.x; i < n; i += stride) {
    int k = (int)(i / N);
    int c = (int)(i % N);
    WT[(long)c * K + k] = (half_t)W[i];
  }
}

__global__ __launch_bounds__(256) void k_zero_f32(float* __restrict__ p, long n) {
  long stride = (long)gridDim.x * blockDim.x;
  for (long i = (long)blockIdx.x * blockDim.x + threadIdx.x; i < n; i += stride)
    p[i] = 0.0f;
}

// t_in[t] = [obj[s[t]] | pred[t] | obj[o[t]]], width Din each. Din % 8 == 0.
__global__ __launch_bounds__(256) void k_gather3(const half_t* __restrict__ obj,
                                                 const half_t* __restrict__ pred,
                                                 const int* __restrict__ s,
                                                 const int* __restrict__ o,
                                                 half_t* __restrict__ out,
                                                 int T, int Din) {
  int W = 3 * Din;
  int cpr = W >> 3;
  long n = (long)T * cpr;
  long stride = (long)gridDim.x * blockDim.x;
  for (long i = (long)blockIdx.x * blockDim.x + threadIdx.x; i < n; i += stride) {
    int t  = (int)(i / cpr);
    int c8 = (int)(i % cpr) << 3;
    const half_t* src;
    if (c8 < Din)           src = obj  + (long)s[t] * Din + c8;
    else if (c8 < 2 * Din)  src = pred + (long)t * Din + (c8 - Din);
    else                    src = obj  + (long)o[t] * Din + (c8 - 2 * Din);
    *(v8h*)(out + (long)t * W + c8) = *(const v8h*)src;
  }
}

// out[r] = [a_f32[r] (Da) | b_f16[r] (Db)] -> f16
__global__ __launch_bounds__(256) void k_concat(const float* __restrict__ a, int Da,
                                                const half_t* __restrict__ b, int Db,
                                                half_t* __restrict__ out, int rows) {
  int w = Da + Db;
  long n = (long)rows * w;
  long stride = (long)gridDim.x * blockDim.x;
  for (long i = (long)blockIdx.x * blockDim.x + threadIdx.x; i < n; i += stride) {
    int r = (int)(i / w);
    int c = (int)(i % w);
    out[i] = (c < Da) ? (half_t)a[(long)r * Da + c] : b[(long)r * Db + (c - Da)];
  }
}

// ---------------------------------------------------------------------------
// Tiled WMMA GEMM:  C(MxN) = act(A(MxK) @ B(KxN) + bias)
//   A  f16 row-major (M x K);  BT f16 row-major (N x K) (B pre-transposed)
// Requires K % 32 == 0 (true for every GEMM in this net).
// Block tile 128x128, 8 waves, 32x64 per wave, K-step 32.
// Full tiles use async global->LDS copies with double buffering when available.
// ---------------------------------------------------------------------------
__global__ __launch_bounds__(256)
void k_gemm_wmma(int M, int N, int K,
                 const half_t* __restrict__ A, int lda,
                 const half_t* __restrict__ BT, int ldbt,
                 const float* __restrict__ bias,
                 half_t* __restrict__ C16, float* __restrict__ C32, int ldc,
                 int act) {
  __shared__ half_t sA[2 * GEMM_BM * LDSS];
  __shared__ half_t sB[2 * GEMM_BN * LDSS];

  const int tid  = threadIdx.x;
  const int lane = tid & 31;
  const int wave = tid >> 5;
  const int wm   = (wave >> 1) * 32;    // 4 waves along M
  const int wn   = (wave & 1) * 64;     // 2 waves along N
  const int m0   = blockIdx.x * GEMM_BM;
  const int n0   = blockIdx.y * GEMM_BN;
  const int l15  = lane & 15;
  const int hi   = lane >> 4;
  const bool fullM = (m0 + GEMM_BM <= M);
  const bool fullN = (n0 + GEMM_BN <= N);

  const int r0 = tid >> 2;              // 0..63
  const int c8 = (tid & 3) << 3;        // 0,8,16,24

  v8f acc[2][4] = {};
  const v8h zero8 = {};

  union Frag { v8h q[2]; v16h h; };

  // compute one staged K-tile from LDS buffers bufA/bufB
  auto compute_tile = [&](const half_t* bufA, const half_t* bufB) {
    Frag fa[2], fb[4];
#pragma unroll
    for (int i = 0; i < 2; ++i) {
      const half_t* base = bufA + (wm + i * 16 + l15) * LDSS + hi * 8;
      fa[i].q[0] = *(const v8h*)(base);
      fa[i].q[1] = *(const v8h*)(base + 16);
    }
#pragma unroll
    for (int j = 0; j < 4; ++j) {
      const half_t* base = bufB + (wn + j * 16 + l15) * LDSS + hi * 8;
      fb[j].q[0] = *(const v8h*)(base);
      fb[j].q[1] = *(const v8h*)(base + 16);
    }
#pragma unroll
    for (int i = 0; i < 2; ++i)
#pragma unroll
      for (int j = 0; j < 4; ++j)
        acc[i][j] = __builtin_amdgcn_wmma_f32_16x16x32_f16(
            false, fa[i].h, false, fb[j].h, (short)0, acc[i][j], false, false);
  };

#if HAVE_ASYNC_LDS
  if (fullM && fullN) {
    // ---- async double-buffered path (ASYNCcnt-tracked, no VGPR staging) ----
    auto issue = [&](int buf, int k0) {
#pragma unroll
      for (int it = 0; it < 2; ++it) {
        int r = r0 + it * 64;
        __builtin_amdgcn_global_load_async_to_lds_b128(
            (as1_v4i_p)(A + (long)(m0 + r) * lda + k0 + c8),
            (as3_v4i_p)&sA[buf * GEMM_BM * LDSS + r * LDSS + c8], 0, 0);
        __builtin_amdgcn_global_load_async_to_lds_b128(
            (as1_v4i_p)(BT + (long)(n0 + r) * ldbt + k0 + c8),
            (as3_v4i_p)&sB[buf * GEMM_BN * LDSS + r * LDSS + c8], 0, 0);
      }
    };
    issue(0, 0);
    __builtin_amdgcn_s_wait_asynccnt(0);
    __syncthreads();
    int buf = 0;
    for (int k0 = 0; k0 < K; k0 += GEMM_BK) {
      if (k0 + GEMM_BK < K) issue(buf ^ 1, k0 + GEMM_BK);
      compute_tile(&sA[buf * GEMM_BM * LDSS], &sB[buf * GEMM_BN * LDSS]);
      __builtin_amdgcn_s_wait_asynccnt(0);
      __syncthreads();
      buf ^= 1;
    }
  } else
#endif
  {
    // ---- fallback path: VGPR-staged b128 copies with edge masking ----
    for (int k0 = 0; k0 < K; k0 += GEMM_BK) {
      v8h va[2], vb[2];
#pragma unroll
      for (int it = 0; it < 2; ++it) {
        int r  = r0 + it * 64;
        int gr = m0 + r;
        bool mok = fullM || gr < M;
        va[it] = *(const v8h*)(A + (long)(mok ? gr : 0) * lda + k0 + c8);
        if (!mok) va[it] = zero8;
        int gn = n0 + r;
        bool nok = fullN || gn < N;
        vb[it] = *(const v8h*)(BT + (long)(nok ? gn : 0) * ldbt + k0 + c8);
        if (!nok) vb[it] = zero8;
      }
#pragma unroll
      for (int it = 0; it < 2; ++it) {
        int r = r0 + it * 64;
        *(v8h*)(&sA[r * LDSS + c8]) = va[it];
        *(v8h*)(&sB[r * LDSS + c8]) = vb[it];
      }
      __syncthreads();

      if (k0 + GEMM_BK < K) {
        int pr = m0 + (tid & 127); if (pr >= M) pr = M - 1;
        __builtin_prefetch(A + (long)pr * lda + k0 + GEMM_BK);
        int pn = n0 + (tid & 127); if (pn >= N) pn = N - 1;
        __builtin_prefetch(BT + (long)pn * ldbt + k0 + GEMM_BK);
      }

      compute_tile(sA, sB);
      __syncthreads();
    }
  }

  // ---- epilogue (C/D layout: VGPR r -> rows r / r+8, lanes -> cols) ----
#pragma unroll
  for (int j = 0; j < 4; ++j) {
    int col = n0 + wn + j * 16 + l15;
    bool cok = fullN || col < N;
    float bv = (bias && cok) ? bias[col] : 0.0f;
#pragma unroll
    for (int i = 0; i < 2; ++i) {
#pragma unroll
      for (int r = 0; r < 8; ++r) {
        int row = m0 + wm + i * 16 + r + hi * 8;
        if (cok && (fullM || row < M)) {
          float v = acc[i][j][r] + bv;
          if (act == 1) v = fmaxf(v, 0.0f);
          if (C16) C16[(long)row * ldc + col] = (half_t)v;
          if (C32) C32[(long)row * ldc + col] = v;
        }
      }
    }
  }
}

// ---------------------------------------------------------------------------
// Segment-mean pooling of t_out (width W = 2H+Dout)
// ---------------------------------------------------------------------------
__global__ __launch_bounds__(256) void k_counts(const int* __restrict__ s,
                                                const int* __restrict__ o,
                                                float* __restrict__ counts, int T) {
  long stride = (long)gridDim.x * blockDim.x;
  for (long i = (long)blockIdx.x * blockDim.x + threadIdx.x; i < T; i += stride) {
    atomAddF(&counts[s[i]], 1.0f);
    atomAddF(&counts[o[i]], 1.0f);
  }
}

__global__ __launch_bounds__(256) void k_scatter_pool(const half_t* __restrict__ tout,
                                                      int W, int H, int Dout,
                                                      const int* __restrict__ s,
                                                      const int* __restrict__ o,
                                                      float* __restrict__ pooled, int T) {
  int cpr = H >> 3;
  long n = (long)T * cpr;
  long stride = (long)gridDim.x * blockDim.x;
  for (long i = (long)blockIdx.x * blockDim.x + threadIdx.x; i < n; i += stride) {
    int t  = (int)(i / cpr);
    int c8 = (int)(i % cpr) << 3;
    v8h vs = *(const v8h*)(tout + (long)t * W + c8);
    v8h vo = *(const v8h*)(tout + (long)t * W + H + Dout + c8);
    float* ps = pooled + (long)s[t] * H + c8;
    float* po = pooled + (long)o[t] * H + c8;
#pragma unroll
    for (int u = 0; u < 8; ++u) {
      atomAddF(ps + u, (float)vs[u]);
      atomAddF(po + u, (float)vo[u]);
    }
  }
}

__global__ __launch_bounds__(256) void k_divide(const float* __restrict__ pooled,
                                                const float* __restrict__ counts,
                                                half_t* __restrict__ out, int O, int H) {
  int cpr = H >> 3;
  long n = (long)O * cpr;
  long stride = (long)gridDim.x * blockDim.x;
  for (long i = (long)blockIdx.x * blockDim.x + threadIdx.x; i < n; i += stride) {
    int r  = (int)(i / cpr);
    int c8 = (int)(i % cpr) << 3;
    float inv = 1.0f / fmaxf(counts[r], 1.0f);
    const float* src = pooled + (long)r * H + c8;
    v8h v;
#pragma unroll
    for (int u = 0; u < 8; ++u) v[u] = (half_t)(src[u] * inv);
    *(v8h*)(out + (long)r * H + c8) = v;
  }
}

__global__ __launch_bounds__(256) void k_slice_pred(const half_t* __restrict__ tout,
                                                    int W, int H, int Dout,
                                                    half_t* __restrict__ p16,
                                                    float* __restrict__ p32, int T) {
  long n = (long)T * Dout;
  long stride = (long)gridDim.x * blockDim.x;
  for (long i = (long)blockIdx.x * blockDim.x + threadIdx.x; i < n; i += stride) {
    int t = (int)(i / Dout);
    int c = (int)(i % Dout);
    float v = (float)tout[(long)t * W + H + c];
    p16[i] = (half_t)v;
    if (p32) p32[i] = v;
  }
}

// ---------------------------------------------------------------------------
// BatchNorm head
// ---------------------------------------------------------------------------
__global__ __launch_bounds__(256) void k_bnstats(const float* __restrict__ h, int T, int C,
                                                 float* __restrict__ sum,
                                                 float* __restrict__ sumsq) {
  int chunks = (T + 511) >> 9;
  long n = (long)chunks * C;
  long stride = (long)gridDim.x * blockDim.x;
  for (long i = (long)blockIdx.x * blockDim.x + threadIdx.x; i < n; i += stride) {
    int c = (int)(i % C);
    int chunk = (int)(i / C);
    int r0 = chunk << 9;
    int r1 = r0 + 512; if (r1 > T) r1 = T;
    float s = 0.0f, q = 0.0f;
    for (int r = r0; r < r1; ++r) {
      float x = h[(long)r * C + c];
      s += x; q += x * x;
    }
    atomAddF(&sum[c], s);
    atomAddF(&sumsq[c], q);
  }
}

__global__ __launch_bounds__(256) void k_bnfinal(float* __restrict__ sum,
                                                 float* __restrict__ sumsq,
                                                 int T, int C, float eps) {
  int c = blockIdx.x * blockDim.x + threadIdx.x;
  if (c >= C) return;
  float mu  = sum[c] / (float)T;
  float var = sumsq[c] / (float)T - mu * mu;
  sum[c]   = mu;
  sumsq[c] = rsqrtf(var + eps);
}

__global__ __launch_bounds__(256) void k_bnapply(const float* __restrict__ h,
                                                 const float* __restrict__ mu,
                                                 const float* __restrict__ rstd,
                                                 const float* __restrict__ gamma,
                                                 const float* __restrict__ beta,
                                                 half_t* __restrict__ out,
                                                 int T, int C, float alpha) {
  long n = (long)T * C;
  long stride = (long)gridDim.x * blockDim.x;
  for (long i = (long)blockIdx.x * blockDim.x + threadIdx.x; i < n; i += stride) {
    int c = (int)(i % C);
    float y = (h[i] - mu[c]) * rstd[c] * gamma[c] + beta[c];
    y = y > 0.0f ? y : alpha * y;
    out[i] = (half_t)y;
  }
}

__global__ __launch_bounds__(256) void k_softmax_argmax(const float* __restrict__ logits,
                                                        float* __restrict__ cls,
                                                        float* __restrict__ idx, int T, int R) {
  long stride = (long)gridDim.x * blockDim.x;
  for (long t = (long)blockIdx.x * blockDim.x + threadIdx.x; t < T; t += stride) {
    float mx = -1e30f; int am = 0;
    for (int r = 0; r < R; ++r) {
      float v = logits[t * R + r];
      if (v > mx) { mx = v; am = r; }
    }
    float s = 0.0f;
    for (int r = 0; r < R; ++r) {
      float e = expf(logits[t * R + r] - mx);
      cls[t * R + r] = e;
      s += e;
    }
    float inv = 1.0f / s;
    for (int r = 0; r < R; ++r) cls[t * R + r] *= inv;
    idx[t] = (float)am;
  }
}

// ---------------------------------------------------------------------------
// Host orchestration
// ---------------------------------------------------------------------------
namespace {

struct Alloc {
  char* p;
  void* take(size_t bytes) {
    void* r = (void*)p;
    p += (bytes + 255) & ~(size_t)255;
    return r;
  }
};

void gemm(hipStream_t st, int M, int N, int K,
          const half_t* A, int lda, const half_t* BT, int ldbt,
          const float* bias, half_t* C16, float* C32, int ldc, int act) {
  dim3 grid((unsigned)cdiv(M, GEMM_BM), (unsigned)cdiv(N, GEMM_BN));
  k_gemm_wmma<<<grid, dim3(256), 0, st>>>(M, N, K, A, lda, BT, ldbt, bias, C16, C32, ldc, act);
}

struct ConvW {
  const half_t *W1, *W2, *W3, *W4;  // f16, TRANSPOSED (N x K)
  const float  *b1, *b2, *b3, *b4;  // f32 biases
};

struct Scratch {
  half_t* tin;
  half_t* hbuf;
  half_t* tout;
  float*  pooled;
  float*  counts;
  half_t* pooledh;
  half_t* objh2;
};

void conv_layer(hipStream_t st, int T, int O, int Din, int H, int Dout,
                const half_t* objIn, const half_t* predIn,
                const int* sidx, const int* oidx,
                const ConvW& w, const Scratch& ws,
                half_t* objOut, half_t* predOut,
                float* objOutF32, float* predOutF32) {
  const int W2N = 2 * H + Dout;
  {
    long n = (long)T * 3 * Din / 8;
    k_gather3<<<dim3((unsigned)cdiv(n, 256)), dim3(256), 0, st>>>(objIn, predIn, sidx, oidx,
                                                                  ws.tin, T, Din);
  }
  gemm(st, T, H, 3 * Din, ws.tin, 3 * Din, w.W1, 3 * Din, w.b1, ws.hbuf, nullptr, H, 1);
  gemm(st, T, W2N, H, ws.hbuf, H, w.W2, H, w.b2, ws.tout, nullptr, W2N, 1);
  {
    long n = (long)T * Dout;
    k_slice_pred<<<dim3((unsigned)cdiv(n, 256)), dim3(256), 0, st>>>(ws.tout, W2N, H, Dout,
                                                                     predOut, predOutF32, T);
  }
  {
    long n = (long)O * H;
    k_zero_f32<<<dim3((unsigned)cdiv(n, 256)), dim3(256), 0, st>>>(ws.pooled, n);
    k_zero_f32<<<dim3((unsigned)cdiv(O, 256)), dim3(256), 0, st>>>(ws.counts, O);
    k_counts<<<dim3((unsigned)cdiv(T, 256)), dim3(256), 0, st>>>(sidx, oidx, ws.counts, T);
    long ns = (long)T * H / 8;
    k_scatter_pool<<<dim3((unsigned)cdiv(ns, 256)), dim3(256), 0, st>>>(ws.tout, W2N, H, Dout,
                                                                        sidx, oidx, ws.pooled, T);
    k_divide<<<dim3((unsigned)cdiv(n / 8, 256)), dim3(256), 0, st>>>(ws.pooled, ws.counts,
                                                                     ws.pooledh, O, H);
  }
  gemm(st, O, H, H, ws.pooledh, H, w.W3, H, w.b3, ws.objh2, nullptr, H, 1);
  gemm(st, O, Dout, H, ws.objh2, H, w.W4, H, w.b4, objOut, objOutF32, Dout, 1);
}

}  // namespace

extern "C" void kernel_launch(void* const* d_in, const int* in_sizes, int n_in,
                              void* d_out, int out_size, void* d_ws, size_t ws_size,
                              hipStream_t stream) {
  (void)in_sizes; (void)n_in; (void)out_size; (void)ws_size;
  const int T = T_N, O = O_N, R = R_N;

  const float* obj_vecs     = (const float*)d_in[0];
  const float* pred_gt_vecs = (const float*)d_in[1];
  const float* pred_vecs    = (const float*)d_in[2];
  const int*   s_idx        = (const int*)d_in[3];
  const int*   o_idx        = (const int*)d_in[4];
  const int GC[4][3] = {{64, 512, 128}, {128, 512, 128}, {128, 512, 128}, {128, 128, 32}};
  const int GP[4][3] = {{96, 512, 128}, {128, 512, 128}, {128, 512, 128}, {128, 128, 128}};

  float* out_cls    = (float*)d_out;
  float* out_newp   = out_cls + (long)T * R;
  float* out_objgt  = out_newp + T;
  float* out_predgt = out_objgt + (long)O * 32;

  Alloc a{(char*)d_ws};
  Scratch ws;
  ws.tin     = (half_t*)a.take((size_t)T * 384 * sizeof(half_t));
  ws.hbuf    = (half_t*)a.take((size_t)T * 512 * sizeof(half_t));
  ws.tout    = (half_t*)a.take((size_t)T * 1152 * sizeof(half_t));
  ws.pooled  = (float*) a.take((size_t)O * 512 * sizeof(float));
  ws.counts  = (float*) a.take((size_t)O * sizeof(float));
  ws.pooledh = (half_t*)a.take((size_t)O * 512 * sizeof(half_t));
  ws.objh2   = (half_t*)a.take((size_t)O * 512 * sizeof(half_t));
  half_t* objA  = (half_t*)a.take((size_t)O * 128 * sizeof(half_t));
  half_t* objB  = (half_t*)a.take((size_t)O * 128 * sizeof(half_t));
  half_t* predA = (half_t*)a.take((size_t)T * 128 * sizeof(half_t));
  half_t* predB = (half_t*)a.take((size_t)T * 128 * sizeof(half_t));
  float*  bnsum = (float*) a.take(512 * sizeof(float));
  float*  bnsq  = (float*) a.take(512 * sizeof(float));
  float*  logits= (float*) a.take((size_t)T * R * sizeof(float));

  ConvW gc[4], gp[4];
  auto cvtT = [&](int idx, int K, int N) -> half_t* {
    half_t* dst = (half_t*)a.take((size_t)K * N * sizeof(half_t));
    long n = (long)K * N;
    k_cvt_transpose<<<dim3((unsigned)cdiv(n, 256)), dim3(256), 0, stream>>>(
        (const float*)d_in[idx], dst, K, N);
    return dst;
  };
  for (int l = 0; l < 4; ++l) {
    int Din = GC[l][0], H = GC[l][1], Dout = GC[l][2];
    int base = 5 + l * 8;
    gc[l].W1 = cvtT(base + 0, 3 * Din, H);
    gc[l].b1 = (const float*)d_in[base + 1];
    gc[l].W2 = cvtT(base + 2, H, 2 * H + Dout);
    gc[l].b2 = (const float*)d_in[base + 3];
    gc[l].W3 = cvtT(base + 4, H, H);
    gc[l].b3 = (const float*)d_in[base + 5];
    gc[l].W4 = cvtT(base + 6, H, Dout);
    gc[l].b4 = (const float*)d_in[base + 7];
  }
  for (int l = 0; l < 4; ++l) {
    int Din = GP[l][0], H = GP[l][1], Dout = GP[l][2];
    int base = 37 + l * 8;
    gp[l].W1 = cvtT(base + 0, 3 * Din, H);
    gp[l].b1 = (const float*)d_in[base + 1];
    gp[l].W2 = cvtT(base + 2, H, 2 * H + Dout);
    gp[l].b2 = (const float*)d_in[base + 3];
    gp[l].W3 = cvtT(base + 4, H, H);
    gp[l].b3 = (const float*)d_in[base + 5];
    gp[l].W4 = cvtT(base + 6, H, Dout);
    gp[l].b4 = (const float*)d_in[base + 7];
  }
  half_t* hW1T = cvtT(69, 128, 512);
  const float* hb1    = (const float*)d_in[70];
  const float* hgamma = (const float*)d_in[71];
  const float* hbeta  = (const float*)d_in[72];
  half_t* hW2T = cvtT(73, 512, R);
  const float* hb2    = (const float*)d_in[74];

  // ---- stack 1 (g_c) ----
  {
    long n = (long)O * 64;
    k_f32_to_f16<<<dim3((unsigned)cdiv(n, 256)), dim3(256), 0, stream>>>(obj_vecs, objA, n);
    long m = (long)T * 64;
    k_f32_to_f16<<<dim3((unsigned)cdiv(m, 256)), dim3(256), 0, stream>>>(pred_gt_vecs, predA, m);
  }
  half_t* objCur = objA;  half_t* objNxt = objB;
  half_t* prCur  = predA; half_t* prNxt  = predB;
  for (int l = 0; l < 4; ++l) {
    bool last = (l == 3);
    conv_layer(stream, T, O, GC[l][0], GC[l][1], GC[l][2], objCur, prCur, s_idx, o_idx,
               gc[l], ws, objNxt, prNxt,
               last ? out_objgt : nullptr, last ? out_predgt : nullptr);
    half_t* t;
    t = objCur; objCur = objNxt; objNxt = t;
    t = prCur;  prCur  = prNxt;  prNxt  = t;
  }

  // ---- stack 2 (g_p) ----
  {
    long n = (long)O * 96;
    k_concat<<<dim3((unsigned)cdiv(n, 256)), dim3(256), 0, stream>>>(obj_vecs, 64, objCur, 32,
                                                                     objNxt, O);
    long m = (long)T * 96;
    k_concat<<<dim3((unsigned)cdiv(m, 256)), dim3(256), 0, stream>>>(pred_vecs, 64, prCur, 32,
                                                                     prNxt, T);
  }
  { half_t* t = objCur; objCur = objNxt; objNxt = t;
    t = prCur;  prCur  = prNxt;  prNxt  = t; }
  for (int l = 0; l < 4; ++l) {
    conv_layer(stream, T, O, GP[l][0], GP[l][1], GP[l][2], objCur, prCur, s_idx, o_idx,
               gp[l], ws, objNxt, prNxt, nullptr, nullptr);
    half_t* t;
    t = objCur; objCur = objNxt; objNxt = t;
    t = prCur;  prCur  = prNxt;  prNxt  = t;
  }

  // ---- head ----
  float* hf32 = (float*)ws.tout;
  gemm(stream, T, 512, 128, prCur, 128, hW1T, 128, hb1, nullptr, hf32, 512, 0);
  k_zero_f32<<<dim3(2), dim3(256), 0, stream>>>(bnsum, 512);
  k_zero_f32<<<dim3(2), dim3(256), 0, stream>>>(bnsq, 512);
  {
    long n = (long)cdiv(T, 512) * 512;
    k_bnstats<<<dim3((unsigned)cdiv(n, 256)), dim3(256), 0, stream>>>(hf32, T, 512, bnsum, bnsq);
  }
  k_bnfinal<<<dim3(2), dim3(256), 0, stream>>>(bnsum, bnsq, T, 512, BN_EPS);
  {
    long n = (long)T * 512;
    k_bnapply<<<dim3((unsigned)cdiv(n, 256)), dim3(256), 0, stream>>>(hf32, bnsum, bnsq, hgamma,
                                                                      hbeta, ws.hbuf, T, 512,
                                                                      LEAKY_A);
  }
  gemm(stream, T, R, 512, ws.hbuf, 512, hW2T, 512, hb2, nullptr, logits, R, 0);
  k_softmax_argmax<<<dim3((unsigned)cdiv(T, 256)), dim3(256), 0, stream>>>(logits, out_cls,
                                                                           out_newp, T, R);
}
